// SelfAttention_70506183131665
// MI455X (gfx1250) — compile-verified, matrix operand order checked by reading, not run on hardware
//
#include <hip/hip_runtime.h>
#include <hip/hip_bf16.h>

typedef __attribute__((ext_vector_type(8)))  __bf16 v8bf;
typedef __attribute__((ext_vector_type(16))) __bf16 v16bf;
typedef __attribute__((ext_vector_type(8)))  float  v8f;

#define BATCH 8
#define TSEQ 2048
#define CDIM 1024
#define NEG_INF_F (-4294967295.0f)

// ---------------------------------------------------------------------------
// Fragment load from an LDS tile stored row-major with ld=64 bf16.
// 16-bit WMMA A/B layout (wave32): lane L -> row/col = L&15, half = L>>4,
// elements 0..7 : k = kk+8h .. kk+8h+7 ; elements 8..15 : k = kk+16+8h ..+7
// ---------------------------------------------------------------------------
__device__ __forceinline__ v16bf frag_ld64(const __bf16* base, int row, int kk, int lane) {
  int m = lane & 15, h = lane >> 4;
  const __bf16* p = base + (row + m) * 64 + kk + 8 * h;
  v8bf lo = *(const v8bf*)p;
  v8bf hi = *(const v8bf*)(p + 16);
  v16bf f;
#pragma unroll
  for (int i = 0; i < 8; ++i) { f[i] = lo[i]; f[i + 8] = hi[i]; }
  return f;
}

// ---------------------------------------------------------------------------
// GEMM: out[M][N] = epilogue(A_f32[M][K] @ Bt_bf16[N][K]^T)
// epilogue: v += bias[n]; if(RELU) relu; v *= dscale[n]
// Block 256 thr = 8 waves (2x4), block tile 128x128, wave tile 64x32, k-step 64.
// ---------------------------------------------------------------------------
template <bool HAS_BIAS, bool HAS_DSCALE, bool RELU, bool OUTF, bool OUTB>
__global__ __launch_bounds__(256) void gemm_f32a(
    const float* __restrict__ A, const __bf16* __restrict__ Bt,
    const float* __restrict__ bias, const float* __restrict__ dscale,
    float* __restrict__ outF, __bf16* __restrict__ outB,
    int M, int N, int K)
{
  __shared__ __bf16 sA[128 * 64];   // 16 KB
  __shared__ __bf16 sB[128 * 64];   // 16 KB
  const int tid = threadIdx.x, lane = tid & 31, w = tid >> 5;
  const int bm = blockIdx.y * 128, bn = blockIdx.x * 128;
  const int wm = (w >> 2) * 64, wn = (w & 3) * 32;
  const int lrow = tid >> 1, lk = (tid & 1) * 32;

  const float*  ap = A  + (size_t)(bm + lrow) * K + lk;
  const __bf16* bp = Bt + (size_t)(bn + lrow) * K + lk;

  v8f acc[4][2];
#pragma unroll
  for (int mi = 0; mi < 4; ++mi)
#pragma unroll
    for (int ni = 0; ni < 2; ++ni)
#pragma unroll
      for (int i = 0; i < 8; ++i) acc[mi][ni][i] = 0.0f;

  for (int k0 = 0; k0 < K; k0 += 64) {
    float4 fv[8];
#pragma unroll
    for (int i = 0; i < 8; ++i) fv[i] = *(const float4*)(ap + k0 + 4 * i);
    v8bf bq[4];
#pragma unroll
    for (int i = 0; i < 4; ++i) bq[i] = *(const v8bf*)(bp + k0 + 8 * i);
    __builtin_prefetch(ap + k0 + 64, 0, 1);
    __builtin_prefetch(bp + k0 + 64, 0, 1);

    v8bf pk[4];
#pragma unroll
    for (int i = 0; i < 4; ++i) {
      pk[i][0] = (__bf16)fv[2 * i].x;     pk[i][1] = (__bf16)fv[2 * i].y;
      pk[i][2] = (__bf16)fv[2 * i].z;     pk[i][3] = (__bf16)fv[2 * i].w;
      pk[i][4] = (__bf16)fv[2 * i + 1].x; pk[i][5] = (__bf16)fv[2 * i + 1].y;
      pk[i][6] = (__bf16)fv[2 * i + 1].z; pk[i][7] = (__bf16)fv[2 * i + 1].w;
    }

    __syncthreads();
#pragma unroll
    for (int i = 0; i < 4; ++i) {
      *(v8bf*)&sA[lrow * 64 + lk + 8 * i] = pk[i];
      *(v8bf*)&sB[lrow * 64 + lk + 8 * i] = bq[i];
    }
    __syncthreads();

#pragma unroll
    for (int kk = 0; kk < 64; kk += 32) {
      v16bf af[4], bfr[2];
#pragma unroll
      for (int mi = 0; mi < 4; ++mi) af[mi] = frag_ld64(sA, wm + mi * 16, kk, lane);
#pragma unroll
      for (int ni = 0; ni < 2; ++ni) bfr[ni] = frag_ld64(sB, wn + ni * 16, kk, lane);
#pragma unroll
      for (int mi = 0; mi < 4; ++mi)
#pragma unroll
        for (int ni = 0; ni < 2; ++ni)
          acc[mi][ni] = __builtin_amdgcn_wmma_f32_16x16x32_bf16(
              false, af[mi], false, bfr[ni], (short)0, acc[mi][ni], false, false);
    }
  }

  const int h = lane >> 4, nn = lane & 15;
#pragma unroll
  for (int ni = 0; ni < 2; ++ni) {
    int n = bn + wn + ni * 16 + nn;
    float dsc = HAS_DSCALE ? dscale[n] : 1.0f;
    float bs  = HAS_BIAS   ? bias[n]   : 0.0f;
#pragma unroll
    for (int mi = 0; mi < 4; ++mi) {
      int m0 = bm + wm + mi * 16 + 8 * h;
#pragma unroll
      for (int r = 0; r < 8; ++r) {
        float v = acc[mi][ni][r] + bs;
        if (RELU) v = fmaxf(v, 0.0f);
        v *= dsc;
        size_t idx = (size_t)(m0 + r) * N + n;
        if (OUTF) outF[idx] = v;
        if (OUTB) outB[idx] = (__bf16)v;
      }
    }
  }
}

// ---------------------------------------------------------------------------
// Batched bf16 GEMM: out[b][M][N] = A[b][M][K] @ Bt[b][N][K]^T
// epilogue: v *= scale; if(HAS_MASK && mask[b][n]==0) v=NEG_INF;
//           if(HAS_ADD) v += addsrc[b][m][n]
// ---------------------------------------------------------------------------
template <bool HAS_ADD, bool HAS_MASK, bool OUTF, bool OUTB>
__global__ __launch_bounds__(256) void gemm_bf16(
    const __bf16* __restrict__ A, const __bf16* __restrict__ Bt,
    const float* __restrict__ addsrc, const float* __restrict__ colmask,
    float* __restrict__ outF, __bf16* __restrict__ outB,
    int M, int N, int K, float scale,
    long long batchA, long long batchB, long long batchOut,
    long long batchAdd, long long batchMask)
{
  __shared__ __bf16 sA[128 * 64];
  __shared__ __bf16 sB[128 * 64];
  const int bz = blockIdx.z;
  const __bf16* Ab = A  + (long long)bz * batchA;
  const __bf16* Bb = Bt + (long long)bz * batchB;

  const int tid = threadIdx.x, lane = tid & 31, w = tid >> 5;
  const int bm = blockIdx.y * 128, bn = blockIdx.x * 128;
  const int wm = (w >> 2) * 64, wn = (w & 3) * 32;
  const int lrow = tid >> 1, lk = (tid & 1) * 32;

  const __bf16* ap = Ab + (size_t)(bm + lrow) * K + lk;
  const __bf16* bp = Bb + (size_t)(bn + lrow) * K + lk;

  v8f acc[4][2];
#pragma unroll
  for (int mi = 0; mi < 4; ++mi)
#pragma unroll
    for (int ni = 0; ni < 2; ++ni)
#pragma unroll
      for (int i = 0; i < 8; ++i) acc[mi][ni][i] = 0.0f;

  for (int k0 = 0; k0 < K; k0 += 64) {
    v8bf aq[4], bq[4];
#pragma unroll
    for (int i = 0; i < 4; ++i) {
      aq[i] = *(const v8bf*)(ap + k0 + 8 * i);
      bq[i] = *(const v8bf*)(bp + k0 + 8 * i);
    }
    __builtin_prefetch(ap + k0 + 64, 0, 1);
    __builtin_prefetch(bp + k0 + 64, 0, 1);

    __syncthreads();
#pragma unroll
    for (int i = 0; i < 4; ++i) {
      *(v8bf*)&sA[lrow * 64 + lk + 8 * i] = aq[i];
      *(v8bf*)&sB[lrow * 64 + lk + 8 * i] = bq[i];
    }
    __syncthreads();

#pragma unroll
    for (int kk = 0; kk < 64; kk += 32) {
      v16bf af[4], bfr[2];
#pragma unroll
      for (int mi = 0; mi < 4; ++mi) af[mi] = frag_ld64(sA, wm + mi * 16, kk, lane);
#pragma unroll
      for (int ni = 0; ni < 2; ++ni) bfr[ni] = frag_ld64(sB, wn + ni * 16, kk, lane);
#pragma unroll
      for (int mi = 0; mi < 4; ++mi)
#pragma unroll
        for (int ni = 0; ni < 2; ++ni)
          acc[mi][ni] = __builtin_amdgcn_wmma_f32_16x16x32_bf16(
              false, af[mi], false, bfr[ni], (short)0, acc[mi][ni], false, false);
    }
  }

  const int h = lane >> 4, nn = lane & 15;
#pragma unroll
  for (int ni = 0; ni < 2; ++ni) {
    int n = bn + wn + ni * 16 + nn;
    float mval = HAS_MASK ? colmask[(long long)bz * batchMask + n] : 1.0f;
#pragma unroll
    for (int mi = 0; mi < 4; ++mi) {
      int m0 = bm + wm + mi * 16 + 8 * h;
#pragma unroll
      for (int r = 0; r < 8; ++r) {
        float v = acc[mi][ni][r] * scale;
        if (HAS_MASK) { if (mval == 0.0f) v = NEG_INF_F; }
        size_t idx = (size_t)(m0 + r) * N + n;
        if (HAS_ADD) v += addsrc[(long long)bz * batchAdd + idx];
        if (OUTF) outF[(long long)bz * batchOut + idx] = v;
        if (OUTB) outB[(long long)bz * batchOut + idx] = (__bf16)v;
      }
    }
  }
}

// ---------------------------------------------------------------------------
// Transpose+convert fp32 [R][Cc] -> bf16 [Cc][R], batched over blockIdx.z.
// ---------------------------------------------------------------------------
__global__ void transpose_to_bf16(const float* __restrict__ in, __bf16* __restrict__ out,
                                  int R, int Cc, long long inBatch, long long outBatch)
{
  __shared__ float tile[32][33];
  const float* src = in + (long long)blockIdx.z * inBatch;
  __bf16* dst = out + (long long)blockIdx.z * outBatch;
  int c0 = blockIdx.x * 32, r0 = blockIdx.y * 32;
  int tx = threadIdx.x, ty = threadIdx.y;  // 32 x 8
#pragma unroll
  for (int i = 0; i < 32; i += 8)
    tile[ty + i][tx] = src[(long long)(r0 + ty + i) * Cc + (c0 + tx)];
  __syncthreads();
#pragma unroll
  for (int i = 0; i < 32; i += 8)
    dst[(long long)(c0 + ty + i) * R + (r0 + tx)] = (__bf16)tile[tx][ty + i];
}

// ---------------------------------------------------------------------------
// Row softmax over T=2048 cols, fold query_mask into P (bf16). 256 thr/row.
// ---------------------------------------------------------------------------
__global__ __launch_bounds__(256) void softmax_qmask(const float* __restrict__ S,
                                                     const float* __restrict__ qmask,
                                                     __bf16* __restrict__ P)
{
  __shared__ float red[256];
  const int row = blockIdx.x, tid = threadIdx.x;
  const float* s = S + (size_t)row * TSEQ;
  __bf16* p = P + (size_t)row * TSEQ;

  float v[8], m = -3.4e38f;
#pragma unroll
  for (int i = 0; i < 8; ++i) { v[i] = s[tid + i * 256]; m = fmaxf(m, v[i]); }
  red[tid] = m; __syncthreads();
  for (int st = 128; st > 0; st >>= 1) { if (tid < st) red[tid] = fmaxf(red[tid], red[tid + st]); __syncthreads(); }
  m = red[0]; __syncthreads();

  float sum = 0.0f;
#pragma unroll
  for (int i = 0; i < 8; ++i) { v[i] = __expf(v[i] - m); sum += v[i]; }
  red[tid] = sum; __syncthreads();
  for (int st = 128; st > 0; st >>= 1) { if (tid < st) red[tid] += red[tid + st]; __syncthreads(); }

  float inv = qmask[row] / red[0];
#pragma unroll
  for (int i = 0; i < 8; ++i) p[tid + i * 256] = (__bf16)(v[i] * inv);
}

// ---------------------------------------------------------------------------
// In-place LayerNorm over C=1024 cols. 256 thr/row, 4 elems/thr.
// ---------------------------------------------------------------------------
__global__ __launch_bounds__(256) void layernorm_inplace(float* __restrict__ X,
                                                         const float* __restrict__ gamma,
                                                         const float* __restrict__ beta)
{
  __shared__ float red[256];
  const int row = blockIdx.x, tid = threadIdx.x;
  float* x = X + (size_t)row * CDIM;

  float v[4], s = 0.0f;
#pragma unroll
  for (int i = 0; i < 4; ++i) { v[i] = x[tid + i * 256]; s += v[i]; }
  red[tid] = s; __syncthreads();
  for (int st = 128; st > 0; st >>= 1) { if (tid < st) red[tid] += red[tid + st]; __syncthreads(); }
  float mu = red[0] * (1.0f / CDIM); __syncthreads();

  s = 0.0f;
#pragma unroll
  for (int i = 0; i < 4; ++i) { float d = v[i] - mu; v[i] = d; s += d * d; }
  red[tid] = s; __syncthreads();
  for (int st = 128; st > 0; st >>= 1) { if (tid < st) red[tid] += red[tid + st]; __syncthreads(); }
  float rstd = rsqrtf(red[0] * (1.0f / CDIM) + 1e-6f);

#pragma unroll
  for (int i = 0; i < 4; ++i) { int c = tid + i * 256; x[c] = v[i] * rstd * gamma[c] + beta[c]; }
}

// ---------------------------------------------------------------------------
extern "C" void kernel_launch(void* const* d_in, const int* in_sizes, int n_in,
                              void* d_out, int out_size, void* d_ws, size_t ws_size,
                              hipStream_t stream)
{
  (void)in_sizes; (void)n_in; (void)out_size; (void)ws_size;
  const float* queries     = (const float*)d_in[0];
  const float* keys        = (const float*)d_in[1];
  const float* key_masks   = (const float*)d_in[2];
  const float* query_masks = (const float*)d_in[3];
  const float* u           = (const float*)d_in[4];
  const float* d_diag      = (const float*)d_in[5];
  const float* W1          = (const float*)d_in[6];
  const float* b1          = (const float*)d_in[7];
  const float* W2          = (const float*)d_in[8];
  const float* b2          = (const float*)d_in[9];
  const float* ln_g        = (const float*)d_in[10];
  const float* ln_b        = (const float*)d_in[11];

  const int Bb = BATCH, T = TSEQ, Cc = CDIM;
  const size_t BTC = (size_t)Bb * T * Cc;   // 16M elems
  const size_t BTT = (size_t)Bb * T * T;    // 33.5M elems

  char* ws = (char*)d_ws;
  size_t off = 0;
  auto take = [&](size_t bytes) -> char* {
    char* p = ws + off;
    off = (off + bytes + 255) & ~(size_t)255;
    return p;
  };
  __bf16* uT  = (__bf16*)take((size_t)Cc * Cc * 2);
  __bf16* W1T = (__bf16*)take((size_t)Cc * Cc * 2);
  __bf16* W2T = (__bf16*)take((size_t)Cc * Cc * 2);
  __bf16* F1  = (__bf16*)take(BTC * 2);          // relu(Q@u)*d, bf16
  __bf16* F2  = (__bf16*)take(BTC * 2);          // relu(K@u), bf16
  __bf16* Kt  = (__bf16*)take(BTC * 2);          // keys^T per batch [C][T], bf16
  float*  S   = (float*)take(BTT * 4);           // scores fp32
  __bf16* P   = (__bf16*)take(BTT * 2);          // probs * qmask, bf16
  float*  Xp  = (float*)take(BTC * 4);           // attn@K + Q, then LN (in-place)
  float*  H   = (float*)S;                       // FFN hidden aliases S (dead after softmax)
  float*  out2 = (float*)d_out + BTC;

  // output[0] = queries (passthrough)
  hipMemcpyAsync(d_out, queries, BTC * sizeof(float), hipMemcpyDeviceToDevice, stream);

  dim3 tb(32, 8);
  transpose_to_bf16<<<dim3(Cc / 32, Cc / 32, 1), tb, 0, stream>>>(u,  uT,  Cc, Cc, 0, 0);
  transpose_to_bf16<<<dim3(Cc / 32, Cc / 32, 1), tb, 0, stream>>>(W1, W1T, Cc, Cc, 0, 0);
  transpose_to_bf16<<<dim3(Cc / 32, Cc / 32, 1), tb, 0, stream>>>(W2, W2T, Cc, Cc, 0, 0);
  transpose_to_bf16<<<dim3(Cc / 32, T / 32, Bb), tb, 0, stream>>>(keys, Kt, T, Cc,
      (long long)T * Cc, (long long)Cc * T);

  // F1 = relu(Q@u) * d_diag ; F2 = relu(K@u)
  gemm_f32a<false, true, true, false, true><<<dim3(Cc / 128, (Bb * T) / 128), 256, 0, stream>>>(
      queries, uT, nullptr, d_diag, nullptr, F1, Bb * T, Cc, Cc);
  gemm_f32a<false, false, true, false, true><<<dim3(Cc / 128, (Bb * T) / 128), 256, 0, stream>>>(
      keys, uT, nullptr, nullptr, nullptr, F2, Bb * T, Cc, Cc);

  // S = (F1 @ F2^T) / sqrt(C), key-masked
  gemm_bf16<false, true, true, false><<<dim3(T / 128, T / 128, Bb), 256, 0, stream>>>(
      F1, F2, nullptr, key_masks, S, nullptr,
      T, T, Cc, 1.0f / 32.0f,
      (long long)T * Cc, (long long)T * Cc, (long long)T * T, 0, (long long)T);

  // P = softmax(S) * query_mask  (bf16)
  softmax_qmask<<<Bb * T, 256, 0, stream>>>(S, query_masks, P);

  // Xp = P @ keys + queries
  gemm_bf16<true, false, true, false><<<dim3(Cc / 128, T / 128, Bb), 256, 0, stream>>>(
      P, Kt, queries, nullptr, Xp, nullptr,
      T, Cc, T, 1.0f,
      (long long)T * T, (long long)Cc * T, (long long)T * Cc, (long long)T * Cc, 0);

  // LayerNorm in place
  layernorm_inplace<<<Bb * T, 256, 0, stream>>>(Xp, ln_g, ln_b);

  // FFN
  gemm_f32a<true, false, true, true, false><<<dim3(Cc / 128, (Bb * T) / 128), 256, 0, stream>>>(
      Xp, W1T, b1, nullptr, H, nullptr, Bb * T, Cc, Cc);
  gemm_f32a<true, false, false, true, false><<<dim3(Cc / 128, (Bb * T) / 128), 256, 0, stream>>>(
      H, W2T, b2, nullptr, out2, nullptr, Bb * T, Cc, Cc);
}
